// ClippedShallowPLRNN_65936337928704
// MI455X (gfx1250) — matrix operand order repaired and checked by reading, not compiled, not run on hardware
//
#include <hip/hip_runtime.h>

// CDNA5 / gfx1250 WMMA types
typedef __attribute__((ext_vector_type(16))) __bf16 v16bf;
typedef __attribute__((ext_vector_type(8)))  __bf16 v8bf;
typedef __attribute__((ext_vector_type(8)))  float  v8f;
typedef __attribute__((ext_vector_type(4)))  float  f4;

#define NS   64    // n_states
#define NH   256   // n_hidden
#define ZBS  72    // Zbf row stride (halfs), mult of 8 -> 16B aligned frags
#define GBS  264   // Gbf row stride (halfs)
#define W2S  72    // W2T row stride (halfs)
#define W1S  264   // W1T row stride (halfs)
#define OBS  72    // OBT row stride (halfs)
#define PS   68    // partial-sum row stride (floats), 272B = 17*16 -> f4 aligned

__device__ __forceinline__ unsigned short f2bf(float x) {
  unsigned int u = __float_as_uint(x);
  unsigned int r = u + 0x7FFFu + ((u >> 16) & 1u);   // round-to-nearest-even
  return (unsigned short)(r >> 16);
}

// B fragment: 16 consecutive bf16 (one matrix column slice), 16B-aligned
__device__ __forceinline__ v16bf ldfragB(const unsigned short* p) {
  union { v16bf v; v8bf h[2]; } u;
  u.h[0] = *(const v8bf*)p;
  u.h[1] = *(const v8bf*)(p + 8);
  return u.v;
}
// A fragment: 16 consecutive bf16 (one matrix row slice), 16B-aligned
__device__ __forceinline__ v16bf ldfragA(const unsigned short* p) {
  union { v16bf v; v8bf h[2]; } u;
  u.h[0] = *(const v8bf*)p;
  u.h[1] = *(const v8bf*)(p + 8);
  return u.v;
}

__global__ __launch_bounds__(256, 1)
void plrnn_scan_kernel(const float* __restrict__ z0, const float* __restrict__ A,
                       const float* __restrict__ W1, const float* __restrict__ W2,
                       const float* __restrict__ h1, const float* __restrict__ h2,
                       const float* __restrict__ OB, const float* __restrict__ Ob,
                       const int* __restrict__ ntp, float* __restrict__ out)
{
  // Weight staging buffer (36864B), recycled as GEMM2 partial buffer (34816B).
  __shared__ __align__(16) unsigned char WBUFraw[256 * W2S * 2];
  unsigned short* WBUF = (unsigned short*)WBUFraw;
  float*          P    = (float*)WBUFraw;           // 8 waves x [16 x PS]
  __shared__ __align__(16) unsigned short Zbf[16 * ZBS];
  __shared__ __align__(16) unsigned short Gbf[16 * GBS];
  __shared__ __align__(16) float Zf32[16 * NS];
  __shared__ __align__(16) float XP[2 * 16 * PS];
  __shared__ __align__(16) float h2s[NH];
  __shared__ __align__(16) float As[NS], h1s[NS], Obs[NS];

  const int tid  = threadIdx.x;
  const int lane = tid & 31;
  const int w    = tid >> 5;        // wave 0..7
  const int n16  = lane & 15;       // N (B,C,D) or M (A) index
  const int lh   = lane >> 4;       // lane half
  const int b0   = blockIdx.x * 16; // batch tile base
  const int NT   = *ntp;

  // ---------- stage constants + state into LDS ----------
  h2s[tid] = h2[tid];
  if (tid < NS) { As[tid] = A[tid]; h1s[tid] = h1[tid]; Obs[tid] = Ob[tid]; }
  for (int i = tid; i < 16 * NS; i += 256) {
    int m = i >> 6, c = i & 63;
    float z = z0[(size_t)(b0 + m) * NS + c];
    Zf32[m * NS + c] = z;
    Zbf[m * ZBS + c] = f2bf(z);
  }
  // W2T[n][k] = bf16(W2[k][n]);  W2 is [64 x 256]
  for (int i = tid; i < NS * NH; i += 256) {
    int k = i >> 8, n = i & 255;
    WBUF[n * W2S + k] = f2bf(W2[i]);
  }
  __syncthreads();

  // GEMM1 B frags (z @ W2): wave w owns hidden cols [32w,32w+32), K=64 (2 ksteps)
  v16bf B1[2][2];
  #pragma unroll
  for (int T = 0; T < 2; ++T) {
    const int Nb = 32 * w + 16 * T;
    #pragma unroll
    for (int ks = 0; ks < 2; ++ks)
      B1[T][ks] = ldfragB(&WBUF[(Nb + n16) * W2S + 32 * ks + 16 * lh]);
  }
  __syncthreads();
  // W1T[n][k] = bf16(W1[k][n]);  W1 is [256 x 64]
  for (int i = tid; i < NH * NS; i += 256) {
    int k = i >> 6, n = i & 63;
    WBUF[n * W1S + k] = f2bf(W1[i]);
  }
  __syncthreads();
  // GEMM2 B frags (g @ W1): wave w owns K slice [32w,32w+32), all 4 N tiles
  v16bf B2[4];
  #pragma unroll
  for (int j = 0; j < 4; ++j)
    B2[j] = ldfragB(&WBUF[(16 * j + n16) * W1S + 32 * w + 16 * lh]);
  __syncthreads();
  // OBT[n][k] = bf16(OB[k][n]);  OB is [64 x 64]
  for (int i = tid; i < NS * NS; i += 256) {
    int k = i >> 6, n = i & 63;
    WBUF[n * OBS + k] = f2bf(OB[i]);
  }
  __syncthreads();
  // GEMM3 (z @ OB): wave w -> N tile (w&3), K half (w>>2)
  const int Nb3 = 16 * (w & 3);
  const int Kb3 = 32 * (w >> 2);
  const v16bf B3 = ldfragB(&WBUF[(Nb3 + n16) * OBS + Kb3 + 16 * lh]);
  const float h2v0 = h2s[32 * w + n16];
  const float h2v1 = h2s[32 * w + 16 + n16];
  __syncthreads();

  const int m4 = tid >> 4;                 // reduce: row 0..15
  const int c4 = (tid & 15) << 2;          // reduce: col base (f4)
  const size_t orow = (size_t)(NT + 1) * NS;

  for (int t = 0; ; ++t) {
    // ---------- observation: x_t = z_t @ OB + Ob ----------
    {
      v16bf a3 = ldfragA(&Zbf[n16 * ZBS + Kb3 + 16 * lh]);
      v8f x = {};
      x = __builtin_amdgcn_wmma_f32_16x16x32_bf16(false, a3, false, B3,
                                                  (short)0, x, false, false);
      float* xp = &XP[(w >> 2) * 16 * PS + (8 * lh) * PS + Nb3 + n16];
      #pragma unroll
      for (int v = 0; v < 8; ++v) xp[v * PS] = x[v];
    }
    __syncthreads();
    {
      f4 xa = *(const f4*)&XP[0 * 16 * PS + m4 * PS + c4];
      f4 xb = *(const f4*)&XP[1 * 16 * PS + m4 * PS + c4];
      f4 ob = *(const f4*)&Obs[c4];
      f4 xv = xa + xb + ob;
      *(f4*)&out[(size_t)(b0 + m4) * orow + (size_t)t * NS + c4] = xv;
    }
    if (t == NT) break;

    // ---------- GEMM1: h = z @ W2 ; g = relu(h+h2)-relu(h) -> Gbf ----------
    {
      v16bf a0 = ldfragA(&Zbf[n16 * ZBS + 0  + 16 * lh]);
      v16bf a1 = ldfragA(&Zbf[n16 * ZBS + 32 + 16 * lh]);
      #pragma unroll
      for (int T = 0; T < 2; ++T) {
        v8f hacc = {};
        hacc = __builtin_amdgcn_wmma_f32_16x16x32_bf16(false, a0, false, B1[T][0],
                                                       (short)0, hacc, false, false);
        hacc = __builtin_amdgcn_wmma_f32_16x16x32_bf16(false, a1, false, B1[T][1],
                                                       (short)0, hacc, false, false);
        const float hb = (T == 0) ? h2v0 : h2v1;
        const int Nb = 32 * w + 16 * T;
        unsigned short* gp = &Gbf[(8 * lh) * GBS + Nb + n16];
        #pragma unroll
        for (int v = 0; v < 8; ++v) {
          float hv = hacc[v];
          float g  = fmaxf(hv + hb, 0.0f) - fmaxf(hv, 0.0f);
          gp[v * GBS] = f2bf(g);
        }
      }
    }
    __syncthreads();

    // ---------- GEMM2: partial z-update = g @ W1 (K split over waves) ----------
    {
      v16bf a2 = ldfragA(&Gbf[n16 * GBS + 32 * w + 16 * lh]);
      #pragma unroll
      for (int j = 0; j < 4; ++j) {
        v8f p = {};
        p = __builtin_amdgcn_wmma_f32_16x16x32_bf16(false, a2, false, B2[j],
                                                    (short)0, p, false, false);
        float* pp = &P[w * 16 * PS + (8 * lh) * PS + 16 * j + n16];
        #pragma unroll
        for (int v = 0; v < 8; ++v) pp[v * PS] = p[v];
      }
    }
    __syncthreads();

    // ---------- reduce partials: z_{t+1} = A*z_t + sum + h1 ----------
    {
      f4 s = {};
      #pragma unroll
      for (int pI = 0; pI < 8; ++pI)
        s += *(const f4*)&P[pI * 16 * PS + m4 * PS + c4];
      f4 zo = *(const f4*)&Zf32[m4 * NS + c4];
      f4 av = *(const f4*)&As[c4];
      f4 hv = *(const f4*)&h1s[c4];
      f4 zn = av * zo + s + hv;
      *(f4*)&Zf32[m4 * NS + c4] = zn;
      unsigned int lo = (unsigned int)f2bf(zn.x) | ((unsigned int)f2bf(zn.y) << 16);
      unsigned int hi = (unsigned int)f2bf(zn.z) | ((unsigned int)f2bf(zn.w) << 16);
      *(unsigned int*)&Zbf[m4 * ZBS + c4]     = lo;
      *(unsigned int*)&Zbf[m4 * ZBS + c4 + 2] = hi;
    }
    __syncthreads();
  }
}

extern "C" void kernel_launch(void* const* d_in, const int* in_sizes, int n_in,
                              void* d_out, int out_size, void* d_ws, size_t ws_size,
                              hipStream_t stream) {
  const float* z0 = (const float*)d_in[0];
  const float* A  = (const float*)d_in[1];
  const float* W1 = (const float*)d_in[2];
  const float* W2 = (const float*)d_in[3];
  const float* h1 = (const float*)d_in[4];
  const float* h2 = (const float*)d_in[5];
  const float* OB = (const float*)d_in[6];
  const float* Ob = (const float*)d_in[7];
  const int*   nt = (const int*)d_in[8];
  float* out = (float*)d_out;

  const int bs = in_sizes[0] / NS;   // 256
  const int nblocks = bs / 16;       // 16 workgroups, one 16-row tile each

  hipLaunchKernelGGL(plrnn_scan_kernel, dim3(nblocks), dim3(256), 0, stream,
                     z0, A, W1, W2, h1, h2, OB, Ob, nt, out);
}
